// ReverseRNNLayer_73074573574766
// MI455X (gfx1250) — compile-verified, matrix operand order checked by reading, not run on hardware
//
#include <hip/hip_runtime.h>
#include <math.h>

// ---------------------------------------------------------------------------
// ReverseRNNLayer for MI455X (gfx1250), fused single-pass design.
//
// Shapes: B=64, S=2048, F=H=1024, chunk L=128, NC=16 chunks.
// Recurrence is row-independent -> each block owns 16 sequences (one batch b)
// and runs the full 128-step reverse scan locally, h kept in LDS (bf16).
// Both GEMMs (x@Wih^T and h@Whh^T) are fused per step using
// v_wmma_f32_16x16x32_bf16 with f32 accumulation. Weights are pre-converted
// to bf16 (4 MB total -> L2 resident, 192 MB L2).
//
// Round 2 change: tanh via hardware v_tanh_f32 (TRANS32, co-executes with the
// matrix pipe) instead of the ocml polynomial -> ~20x less epilogue VALU.
// ---------------------------------------------------------------------------

typedef __attribute__((ext_vector_type(4)))  __bf16 v4bf;
typedef __attribute__((ext_vector_type(8)))  __bf16 v8bf;
typedef __attribute__((ext_vector_type(16))) __bf16 v16bf;
typedef __attribute__((ext_vector_type(8)))  float  v8f;
typedef __attribute__((ext_vector_type(4)))  float  f32x4;

constexpr int Bc = 64;    // batch
constexpr int Sc = 2048;  // sequence
constexpr int Fc = 1024;  // input features
constexpr int Hc = 1024;  // hidden
constexpr int Lc = 128;   // chunk length
constexpr int NCc = 16;   // number of chunks
constexpr int XS = 1032;  // padded LDS row stride (bf16 elems): 2064B = 516 dw
                          // => 16 rows hit distinct 4-bank groups for b128 ds

// Hardware tanh (CDNA5 transcendental VALU; tracked as TRANS32).
static __device__ __forceinline__ float htanh(float x) {
  float r;
  asm("v_tanh_f32 %0, %1" : "=v"(r) : "v"(x));
  return r;
}

// Load a 16-element bf16 operand in the CDNA5 16-bit 16x32 WMMA lane layout.
// p already includes the per-lane (laneHi*8) K offset; halves are K and K+16.
static __device__ __forceinline__ v16bf load_tile16(const __bf16* p) {
  v8bf lo = *(const v8bf*)(p);
  v8bf hi = *(const v8bf*)(p + 16);
  v16bf r;
#pragma unroll
  for (int i = 0; i < 8; ++i) { r[i] = lo[i]; r[i + 8] = hi[i]; }
  return r;
}

// ---- f32 -> bf16 weight conversion (one-time, 8 MB read / 4 MB write) -----
__global__ void cvt_bf16_kernel(const float* __restrict__ in,
                                __bf16* __restrict__ out, int n) {
  int i = (blockIdx.x * blockDim.x + threadIdx.x) * 4;
  if (i + 3 < n) {
    f32x4 f = *(const f32x4*)(in + i);
    v4bf o;
#pragma unroll
    for (int q = 0; q < 4; ++q) o[q] = (__bf16)f[q];
    *(v4bf*)(out + i) = o;
  }
}

// ---- fused reverse-chunk RNN scan -----------------------------------------
__launch_bounds__(1024, 1)
__global__ void rnn_scan_kernel(const float* __restrict__ x,     // (B,S,F)
                                const __bf16* __restrict__ wih,  // (H,F) bf16
                                const __bf16* __restrict__ whh,  // (H,H) bf16
                                const float* __restrict__ bih,   // (H)
                                const float* __restrict__ bhh,   // (H)
                                float* __restrict__ ys,          // (B,S,H)
                                float* __restrict__ hout)        // (B,H)
{
  __shared__ alignas(16) __bf16 hbuf[16 * XS];  // hidden state, bf16
  __shared__ alignas(16) __bf16 xbuf[16 * XS];  // staged x_t, bf16

  const int tid  = threadIdx.x;
  const int wave = tid >> 5;
  const int lane = tid & 31;
  const int m16  = lane & 15;   // tile row (A/B) == output column (C/D)
  const int hi8  = (lane >> 4) * 8;
  const int b    = blockIdx.x;  // one batch index per block; rows c = 0..15

  const int n0 = wave * 32;     // this wave's first output column (2 N-tiles)

  // bias (b_ih + b_hh) per output column, constant over time
  const float bias0 = bih[n0 + m16]      + bhh[n0 + m16];
  const float bias1 = bih[n0 + 16 + m16] + bhh[n0 + 16 + m16];

  // weight row pointers for the two N-tiles (row-major, K contiguous);
  // out = A @ W^T  =>  "B^T" rows are exactly W rows.
  const __bf16* wih0 = wih + (size_t)(n0 + m16) * Fc + hi8;
  const __bf16* wih1 = wih + (size_t)(n0 + 16 + m16) * Fc + hi8;
  const __bf16* whh0 = whh + (size_t)(n0 + m16) * Hc + hi8;
  const __bf16* whh1 = whh + (size_t)(n0 + 16 + m16) * Hc + hi8;

  // h0 = 0 for every chunk (reference zeroes the carry per chunk)
  for (int i = tid; i < 16 * XS; i += 1024) hbuf[i] = (__bf16)0.0f;

  const __bf16* hrow = &hbuf[m16 * XS + hi8];
  const __bf16* xrow = &xbuf[m16 * XS + hi8];

#pragma unroll 1
  for (int t = 0; t < Lc; ++t) {
    // ---- stage x_t (f32 -> bf16) into LDS; row c reads time c*L + (L-1-t)
    {
      const int e0  = tid * 16;        // 16 rows * 1024 cols / 1024 threads
      const int row = e0 >> 10;
      const int col = e0 & 1023;
      const float* src =
          x + ((size_t)b * Sc + (size_t)row * Lc + (Lc - 1 - t)) * Fc + col;
      const f32x4* s4 = (const f32x4*)src;
#pragma unroll
      for (int q = 0; q < 2; ++q) {
        f32x4 f0 = s4[q * 2 + 0];
        f32x4 f1 = s4[q * 2 + 1];
        v8bf o;
#pragma unroll
        for (int e = 0; e < 4; ++e) { o[e] = (__bf16)f0[e]; o[4 + e] = (__bf16)f1[e]; }
        *(v8bf*)&xbuf[row * XS + col + q * 8] = o;
      }
      if (t + 1 < Lc) __builtin_prefetch(src - Fc, 0, 0);  // next step's row
    }
    __syncthreads();  // also orders previous step's h writes before reads

    // ---- acc = bias; acc += x_t @ Wih^T + h @ Whh^T  (K = 1024, bf16 WMMA)
    v8f acc0, acc1;
#pragma unroll
    for (int i = 0; i < 8; ++i) { acc0[i] = bias0; acc1[i] = bias1; }

#pragma unroll 4
    for (int k0 = 0; k0 < Fc; k0 += 32) {
      v16bf ax = load_tile16(xrow + k0);  // shared A tiles from LDS
      v16bf ah = load_tile16(hrow + k0);

      v16bf bi0 = load_tile16(wih0 + k0);  // W tiles stream from L2
      v16bf bh0 = load_tile16(whh0 + k0);
      acc0 = __builtin_amdgcn_wmma_f32_16x16x32_bf16(
          false, ax, false, bi0, (short)0, acc0, false, false);
      acc0 = __builtin_amdgcn_wmma_f32_16x16x32_bf16(
          false, ah, false, bh0, (short)0, acc0, false, false);

      v16bf bi1 = load_tile16(wih1 + k0);
      v16bf bh1 = load_tile16(whh1 + k0);
      acc1 = __builtin_amdgcn_wmma_f32_16x16x32_bf16(
          false, ax, false, bi1, (short)0, acc1, false, false);
      acc1 = __builtin_amdgcn_wmma_f32_16x16x32_bf16(
          false, ah, false, bh1, (short)0, acc1, false, false);
    }
    __syncthreads();  // all waves done reading hbuf before anyone rewrites it

    // ---- tanh, write back h (LDS bf16) and ys (global f32)
    const int l = Lc - 1 - t;  // ascending chunk-local time of this output
#pragma unroll
    for (int v = 0; v < 8; ++v) {
      const int m = v + hi8;  // sequence-in-block (chunk index c)
      float h0 = htanh(acc0[v]);
      float h1 = htanh(acc1[v]);
      hbuf[m * XS + n0 + m16]      = (__bf16)h0;
      hbuf[m * XS + n0 + 16 + m16] = (__bf16)h1;
      // ys[b, (NC-1-c)*L + l, :]
      const size_t so =
          ((size_t)b * Sc + (size_t)(NCc - 1 - m) * Lc + l) * (size_t)Hc;
      ys[so + n0 + m16]      = h0;
      ys[so + n0 + 16 + m16] = h1;
      // h_final = carry after last step for chunk c = NC-1
      if (t == Lc - 1 && m == NCc - 1) {
        hout[(size_t)b * Hc + n0 + m16]      = h0;
        hout[(size_t)b * Hc + n0 + 16 + m16] = h1;
      }
    }
  }
}

extern "C" void kernel_launch(void* const* d_in, const int* in_sizes, int n_in,
                              void* d_out, int out_size, void* d_ws,
                              size_t ws_size, hipStream_t stream) {
  const float* x   = (const float*)d_in[0];
  // d_in[1] = hidden_state (zeros; the reference re-zeroes per chunk anyway)
  const float* wih = (const float*)d_in[2];
  const float* whh = (const float*)d_in[3];
  const float* bih = (const float*)d_in[4];
  const float* bhh = (const float*)d_in[5];

  __bf16* wih_bf = (__bf16*)d_ws;                   // 2 MB
  __bf16* whh_bf = wih_bf + (size_t)Hc * Fc;        // 2 MB

  {
    const int n = Hc * Fc;  // 1M elements each
    cvt_bf16_kernel<<<n / (256 * 4), 256, 0, stream>>>(wih, wih_bf, n);
    cvt_bf16_kernel<<<n / (256 * 4), 256, 0, stream>>>(whh, whh_bf, n);
  }

  float* ys   = (float*)d_out;
  float* hout = ys + (size_t)Bc * Sc * Hc;

  rnn_scan_kernel<<<Bc, 1024, 0, stream>>>(x, wih_bf, whh_bf, bih, bhh, ys,
                                           hout);
}
